// VectorQuantizer_60567628808414
// MI455X (gfx1250) — compile-verified, matrix operand order checked by reading, not run on hardware
//
#include <hip/hip_runtime.h>

typedef __attribute__((ext_vector_type(16))) _Float16 v16h;
typedef __attribute__((ext_vector_type(8)))  _Float16 v8h;
typedef __attribute__((ext_vector_type(8)))  float    v8f;
typedef __attribute__((ext_vector_type(4)))  int      v4i;

#define K_CB    4096
#define D       256
#define N_TOT   32768          // 32 * 32 * 32 rows
#define HW      1024           // 32 * 32
#define Z_ELEMS 8388608        // 32 * 256 * 32 * 32

#define STAGE_TILES 4          // 16-entry N-tiles per LDS stage
#define STAGE_ROWS  64         // codebook rows per stage (4 * 16)
#define N_STAGES    (K_CB / STAGE_ROWS)   // 64

#define GAS __attribute__((address_space(1)))
#define LAS __attribute__((address_space(3)))

union V16U { v16h v; v8h h[2]; _Float16 e[16]; };

// ---------------------------------------------------------------------------
// Async global->LDS copy of one 16-byte chunk (ASYNCcnt-tracked on CDNA5).
// Builtin signature (from clang diagnostic): (AS1 v4i*, AS3 v4i*, imm, imm).
// Falls back to a synchronous LDS fill if the builtin is unavailable.
// ---------------------------------------------------------------------------
__device__ __forceinline__ void async_b128(const _Float16* gsrc, _Float16* ldst) {
#if __has_builtin(__builtin_amdgcn_global_load_async_to_lds_b128)
    __builtin_amdgcn_global_load_async_to_lds_b128(
        (GAS v4i*)(void*)gsrc, (LAS v4i*)(void*)ldst, 0, 0);
#else
    *(v8h*)ldst = *(const v8h*)gsrc;
#endif
}

__device__ __forceinline__ void wait_async0() {
#if __has_builtin(__builtin_amdgcn_global_load_async_to_lds_b128)
# if __has_builtin(__builtin_amdgcn_s_wait_asynccnt)
    __builtin_amdgcn_s_wait_asynccnt(0);
# else
    asm volatile("s_wait_asynccnt 0x0" ::: "memory");
# endif
#endif
}

// ---------------------------------------------------------------------------
// Kernel 1: codebook f32 -> f16, plus row squared norms e_sq[k].
// ---------------------------------------------------------------------------
__global__ void vq_prep_codebook(const float* __restrict__ cb,
                                 _Float16* __restrict__ cb_h,
                                 float* __restrict__ e_sq) {
    __shared__ float red[256];
    const int row = blockIdx.x;
    const int t   = threadIdx.x;
    float v = cb[row * D + t];
    cb_h[row * D + t] = (_Float16)v;
    red[t] = v * v;
    __syncthreads();
    for (int s = 128; s > 0; s >>= 1) {
        if (t < s) red[t] += red[t + s];
        __syncthreads();
    }
    if (t == 0) e_sq[row] = red[0];
}

// ---------------------------------------------------------------------------
// Kernel 2: fused distance-GEMM + argmin.
// 8 waves/WG; each wave owns a register-resident 16-row A tile. Codebook is
// double-buffered through LDS in 32 KB stages via async global->LDS copies
// (fetched from L2 once per WG instead of once per wave), XOR-swizzled so the
// per-lane ds_load_b128 B fetches are bank-conflict-free.
// ---------------------------------------------------------------------------
__global__ void __launch_bounds__(256)
vq_argmin_wmma(const float* __restrict__ z,
               const _Float16* __restrict__ cb_h,
               const float* __restrict__ e_sq,
               int* __restrict__ idx_out,
               float* __restrict__ idx_out_f) {
    __shared__ _Float16 lds_b[2][STAGE_ROWS * D];   // 2 x 32 KB

    const int tid     = threadIdx.x;
    const int lane    = tid & 31;
    const int wave    = tid >> 5;
    const int rowbase = blockIdx.x * 128 + wave * 16;
    const int m       = lane & 15;   // A-matrix row this lane carries
    const int kgrp    = lane >> 4;   // K-chunk half (per ISA 16-bit A layout)

    // z_flat row n -> NCHW address: z[((b*256 + c)*32 + h)*32 + w]
    const int  n_row = rowbase + m;
    const int  b     = n_row >> 10;
    const int  hw    = n_row & 1023;
    const long zbase = (long)b * (D * HW) + hw;

    // Register-resident A tile (16 rows x 256 dims, f16), loaded once.
    V16U a[8];
#pragma unroll
    for (int ks = 0; ks < 8; ++ks) {
        const int c0 = ks * 32 + kgrp * 8;
#pragma unroll
        for (int j = 0; j < 8; ++j) {
            a[ks].e[j]     = (_Float16)z[zbase + (long)(c0 + j) * HW];
            a[ks].e[8 + j] = (_Float16)z[zbase + (long)(c0 + 16 + j) * HW];
        }
    }

    float best[8];
    int   bidx[8];
#pragma unroll
    for (int v = 0; v < 8; ++v) { best[v] = 3.0e38f; bidx[v] = 0; }

    const int ncol = lane & 15;      // B/C column (codebook entry within tile)

    // ---- prologue: stage 0 into buffer 0 ----
    {
        const _Float16* gs = cb_h;
        _Float16* ld = &lds_b[0][0];
#pragma unroll
        for (int i = 0; i < 8; ++i) {
            const int q   = i * 256 + tid;     // b128 chunk id, 0..2047
            const int row = q >> 5;
            const int chs = (q & 31) ^ (row & 15);   // bank-spread swizzle
            async_b128(gs + q * 8, ld + row * D + chs * 8);
        }
    }
    wait_async0();
    __syncthreads();

    for (int s = 0; s < N_STAGES; ++s) {
        const int buf = s & 1;

        // Issue next stage into the other buffer; overlaps with compute below.
        if (s + 1 < N_STAGES) {
            const _Float16* gs = cb_h + (size_t)(s + 1) * (STAGE_ROWS * D);
            _Float16* ld = &lds_b[buf ^ 1][0];
#pragma unroll
            for (int i = 0; i < 8; ++i) {
                const int q   = i * 256 + tid;
                const int row = q >> 5;
                const int chs = (q & 31) ^ (row & 15);
                async_b128(gs + q * 8, ld + row * D + chs * 8);
            }
        }

#pragma unroll
        for (int ti = 0; ti < STAGE_TILES; ++ti) {
            const int   nt   = s * STAGE_TILES + ti;
            const float esq  = e_sq[nt * 16 + ncol];
            const int   lrow = ti * 16 + ncol;
            const int   sw   = lrow & 15;

            v8f acc = {0.f, 0.f, 0.f, 0.f, 0.f, 0.f, 0.f, 0.f};
#pragma unroll
            for (int ks = 0; ks < 8; ++ks) {
                const int ch0 = ks * 4 + kgrp;       // 8-half chunk index
                V16U bu;
                bu.h[0] = *(const v8h*)&lds_b[buf][lrow * D + ((ch0)     ^ sw) * 8];
                bu.h[1] = *(const v8h*)&lds_b[buf][lrow * D + ((ch0 + 2) ^ sw) * 8];
                acc = __builtin_amdgcn_wmma_f32_16x16x32_f16(
                    false, a[ks].v, false, bu.v, (short)0, acc, false, false);
            }

            // dist(m,n) = e_sq[n] - 2*<z_m, e_n>  (||z||^2 is argmin-invariant)
#pragma unroll
            for (int v = 0; v < 8; ++v) {
                const float d = esq - 2.0f * acc[v];
                if (d < best[v]) { best[v] = d; bidx[v] = nt * 16 + ncol; }
            }
        }

        wait_async0();      // next stage resident in LDS
        __syncthreads();    // all waves done reading current buffer
    }

    // C layout: slot (VGPR v, lane) holds row m = v + 8*(lane>>4), col = lane&15.
    // Reduce min+argmin across the 16 lanes of each row; lowest index on tie.
#pragma unroll
    for (int v = 0; v < 8; ++v) {
        float d  = best[v];
        int   bi = bidx[v];
        for (int off = 8; off > 0; off >>= 1) {
            const float od = __shfl_xor(d, off, 16);
            const int   oi = __shfl_xor(bi, off, 16);
            if (od < d || (od == d && oi < bi)) { d = od; bi = oi; }
        }
        if (m == 0) {  // lanes 0 and 16 publish rows v and v+8
            const int row = rowbase + v + 8 * kgrp;
            idx_out[row]   = bi;
            idx_out_f[row] = (float)bi;
        }
    }
}

// ---------------------------------------------------------------------------
// Kernel 3: gather z_q = codebook[idx] into NCHW (coalesced writes), plus
// per-block partial sums of (z_q - z)^2.  1024 blocks (b,h) x 256 threads.
// ---------------------------------------------------------------------------
__global__ void vq_gather_loss(const float* __restrict__ z,
                               const float* __restrict__ cb,
                               const int* __restrict__ idx,
                               float* __restrict__ zq_out,
                               float* __restrict__ partials) {
    __shared__ float red[256];
    const int bh = blockIdx.x;           // b*32 + h
    const int b  = bh >> 5, h = bh & 31;
    const int t  = threadIdx.x;
    const int w  = t & 31;
    const int c0 = t >> 5;
    const int n  = b * 1024 + h * 32 + w;

    const int my = idx[n];
    const float* crow = cb + (size_t)my * D;
    const long obase  = (long)b * (D * HW) + h * 32 + w;

    float ssd = 0.f;
    for (int c = c0; c < D; c += 8) {
        const float q = crow[c];
        const long  o = obase + (long)c * HW;
        zq_out[o] = q;                   // z_q_st == z_q (straight-through)
        const float df = q - z[o];
        ssd += df * df;
    }
    red[t] = ssd;
    __syncthreads();
    for (int s = 128; s > 0; s >>= 1) {
        if (t < s) red[t] += red[t + s];
        __syncthreads();
    }
    if (t == 0) partials[bh] = red[0];
}

// ---------------------------------------------------------------------------
// Kernel 4: deterministic fixed-order reduction of 1024 partials -> vq_loss.
// codebook_loss == commitment_loss numerically, so loss = 1.25 * SSD / numel.
// ---------------------------------------------------------------------------
__global__ void vq_finalize(const float* __restrict__ partials,
                            float* __restrict__ loss_out) {
    __shared__ float red[256];
    const int t = threadIdx.x;
    red[t] = partials[t] + partials[t + 256] + partials[t + 512] + partials[t + 768];
    __syncthreads();
    for (int s = 128; s > 0; s >>= 1) {
        if (t < s) red[t] += red[t + s];
        __syncthreads();
    }
    if (t == 0) *loss_out = red[0] * (1.25f / (float)Z_ELEMS);
}

// ---------------------------------------------------------------------------
extern "C" void kernel_launch(void* const* d_in, const int* in_sizes, int n_in,
                              void* d_out, int out_size, void* d_ws, size_t ws_size,
                              hipStream_t stream) {
    const float* z  = (const float*)d_in[0];   // (32,256,32,32) f32
    const float* cb = (const float*)d_in[1];   // (4096,256) f32

    float* out  = (float*)d_out;
    float* zq   = out;                  // 8388608 floats
    float* loss = out + Z_ELEMS;        // 1 float
    float* idxf = out + Z_ELEMS + 1;    // 32768 floats (indices)

    char* ws = (char*)d_ws;
    _Float16* cb_h     = (_Float16*)ws;                                   // 2 MB
    float*    e_sq     = (float*)(ws + 2u * 1024u * 1024u);               // 16 KB
    int*      idx      = (int*)  (ws + 2u * 1024u * 1024u + 16u * 1024u); // 128 KB
    float*    partials = (float*)(ws + 2u * 1024u * 1024u + 16u * 1024u
                                     + 128u * 1024u);                     // 4 KB

    vq_prep_codebook<<<K_CB, 256, 0, stream>>>(cb, cb_h, e_sq);
    vq_argmin_wmma<<<N_TOT / 128, 256, 0, stream>>>(z, cb_h, e_sq, idx, idxf);
    vq_gather_loss<<<1024, 256, 0, stream>>>(z, cb, idx, zq, partials);
    vq_finalize<<<1, 256, 0, stream>>>(partials, loss);
}
